// MRNN_57887569216036
// MI455X (gfx1250) — matrix operand order repaired and checked by reading, not compiled, not run on hardware
//
#include <hip/hip_runtime.h>
#include <hip/hip_bf16.h>
#include <cstdint>
#include <cstddef>

// ---------------------------------------------------------------------------
// mRNN (fractional-memory RNN), T=256 B=64 IN=32 H=256 OUT=32 K=16, 2 layers.
// Persistent single-WGP kernel: all per-step GEMMs via v_wmma_f32_16x16x32_bf16,
// activations LDS-resident, bf16 weights streamed from L2, K=16 hidden-state
// history kept as lane-private circular buffers in workspace (same lane writes
// and reads each (b,h) slot, so no cross-lane global coherence is needed).
// Register-pressure tuned: single accumulator set live at a time, depth-1
// double buffering of weight fragments, and inline-asm pointer laundering to
// stop LICM from hoisting (and then spilling) the t-invariant weight loads.
// ---------------------------------------------------------------------------

typedef __attribute__((ext_vector_type(16))) __bf16 v16bf;
typedef __attribute__((ext_vector_type(8)))  __bf16 v8bf;
typedef __attribute__((ext_vector_type(8)))  float  v8f;

namespace {
constexpr int T_ = 256, B_ = 64, IN_ = 32, H_ = 256, OUT_ = 32;
constexpr int PH = 264;            // LDS row pitch (bf16) for H-wide tiles: 256+8 pad
constexpr int PX = 40;             // LDS row pitch (bf16) for x tile: 32+8 pad
constexpr int NTHREADS = 512;      // 16 waves (wave32)

// packed bf16 weight offsets (elements) inside workspace
constexpr int O_WDX0 = 0;
constexpr int O_WX0  = O_WDX0 + H_ * IN_;   // 8192
constexpr int O_WDH0 = O_WX0  + H_ * IN_;   // 16384
constexpr int O_WH0  = O_WDH0 + H_ * H_;    // 81920
constexpr int O_WO0  = O_WH0  + H_ * H_;    // 147456
constexpr int O_WDX1 = O_WO0  + H_ * H_;    // 212992
constexpr int O_WDH1 = O_WDX1 + H_ * H_;    // 278528
constexpr int O_WX1  = O_WDH1 + H_ * H_;    // 344064
constexpr int O_WH1  = O_WX1  + H_ * H_;    // 409600
constexpr int O_WO1  = O_WH1  + H_ * H_;    // 475136
constexpr int BUF_BYTE_OFF = 1 << 20;       // history buffers start at 1MB
}

__device__ __forceinline__ v8f zero8() {
  v8f z = {0.f, 0.f, 0.f, 0.f, 0.f, 0.f, 0.f, 0.f};
  return z;
}

// Opaque copy of a uniform pointer (kept in an SGPR pair). Defeats LICM so
// weight loads stay at their use site inside the persistent loop instead of
// being hoisted out and spilled to scratch.
__device__ __forceinline__ const __bf16* launder(const __bf16* p) {
  asm volatile("" : "+s"(p));
  return p;
}

__device__ __forceinline__ v8f wmma_bf16(v16bf a, v16bf b, v8f c) {
  return __builtin_amdgcn_wmma_f32_16x16x32_bf16(false, a, false, b, (short)0, c,
                                                 false, false);
}

// A-fragment (16x32 bf16) from an LDS activation tile, contiguous k order (same as load_b):
// lane L: row = L%16; elems e -> K = kb + (L/16)*16 + e
__device__ __forceinline__ v16bf load_a(const __bf16* base, int pitch, int mtile,
                                        int kb, int lrow, int lhi) {
  const __bf16* p = base + (mtile * 16 + lrow) * pitch + kb + lhi * 16;
  v8bf lo = *(const v8bf*)p;
  v8bf hi = *(const v8bf*)(p + 8);
  v16bf a;
#pragma unroll
  for (int i = 0; i < 8; ++i) { a[i] = lo[i]; a[i + 8] = hi[i]; }
  return a;
}

// B-fragment (32x16 bf16) for D = A x W^T with W row-major [N, Kdim]:
// lane L: col n = nt*16 + L%16; elems e -> K = kb + (L/16)*16 + e (contiguous 32B)
__device__ __forceinline__ v16bf load_b(const __bf16* W, int kdim, int nt, int kb,
                                        int lrow, int lhi) {
  return *(const v16bf*)(W + (nt * 16 + lrow) * kdim + kb + lhi * 16);
}

// K-loop with depth-1 double buffering of the weight fragment; sched_barrier
// keeps the MI scheduler from hoisting all 8 B-fragments at once.
template <int KSTEPS>
__device__ __forceinline__ void gemm_acc(v8f (&c)[4], const __bf16* A, int pitch,
                                         const __bf16* W, int kdim, int ntile,
                                         int lrow, int lhi) {
  W = launder(W);
  v16bf b = load_b(W, kdim, ntile, 0, lrow, lhi);
#pragma unroll
  for (int ks = 0; ks < KSTEPS; ++ks) {
    const int kb = ks * 32;
    v16bf bn;
    if (ks + 1 < KSTEPS) bn = load_b(W, kdim, ntile, kb + 32, lrow, lhi);
#pragma unroll
    for (int m = 0; m < 4; ++m) {
      v16bf a = load_a(A, pitch, m, kb, lrow, lhi);
      c[m] = wmma_bf16(a, b, c[m]);
    }
    if (ks + 1 < KSTEPS) b = bn;
    __builtin_amdgcn_sched_barrier(0);
  }
}

// d = 0.5*sigmoid(pre+bias); truncated fractional-memory sum over K=16 history
// slots, written straight to an LDS activation tile (A-matrix of next GEMM).
// buf layout: [slot][b][h] bf16; slot of h_{t'} is (t' & 15); slots for t' < 0
// are still zero-initialized (slot s is first written at step s).
__device__ __forceinline__ void d_and_memory(const v8f (&c)[4], float bias,
                                             const __bf16* __restrict__ buf, int t,
                                             int lhi, int hcol,
                                             __bf16* __restrict__ dst) {
#pragma unroll
  for (int m = 0; m < 4; ++m) {
#pragma unroll
    for (int i = 0; i < 8; ++i) {
      const int brow = m * 16 + lhi * 8 + i;
      const float d = 0.5f / (1.0f + __expf(-(c[m][i] + bias)));
      const __bf16* bp = buf + brow * H_ + hcol;
      float w = d;  // pi_1 = d
      float acc = w * (float)bp[((t - 1) & 15) * (B_ * H_)];
#pragma unroll
      for (int j = 1; j < 16; ++j) {  // pi_{j+1} = pi_j * (j - d) / (j + 1)
        w *= ((float)j - d) * (1.0f / (float)(j + 1));
        acc += w * (float)bp[((t - 1 - j) & 15) * (B_ * H_)];
      }
      dst[brow * PH + hcol] = (__bf16)acc;
    }
  }
}

__global__ void __launch_bounds__(NTHREADS, 1) mrnn_persistent(
    const float* __restrict__ xg,                        // [T,B,IN]
    const float* __restrict__ bd0, const float* __restrict__ bh0,
    const float* __restrict__ bo0, const float* __restrict__ bd1,
    const float* __restrict__ bh1, const float* __restrict__ bo1,
    const __bf16* __restrict__ wts,                      // packed bf16 weights
    __bf16* __restrict__ buf0, __bf16* __restrict__ buf1,// [16,B,H] history
    float* __restrict__ out) {                           // [T,B,OUT]
  __shared__ __align__(32) __bf16 xb[B_ * PX];    // x_t           (bf16)
  __shared__ __align__(32) __bf16 h0[B_ * PH];    // h_{t-1} L0    (bf16)
  __shared__ __align__(32) __bf16 h1[B_ * PH];    // h_{t-1} L1    (bf16)
  __shared__ __align__(32) __bf16 mb[B_ * PH];    // shared m0 / o0 buffer
  __shared__ __align__(32) __bf16 m1b[B_ * PH];   // m1 buffer

  const int tid = threadIdx.x;
  const int lane = tid & 31, wave = tid >> 5;
  const int lrow = lane & 15, lhi = lane >> 4;
  const int ntile = wave;                 // 16 waves x 16 hidden cols
  const int hcol = ntile * 16 + lrow;     // this wave's hidden column

  // hoisted per-lane biases (bias index depends only on the fixed hcol)
  const float bd0v = bd0[hcol], bh0v = bh0[hcol], bo0v = bo0[hcol];
  const float bd1v = bd1[hcol], bh1v = bh1[hcol];
  const int nt1 = (wave >> 2) & 1;        // output GEMM tiling (waves 0..7)
  const int om = wave & 3;
  const int ocol = nt1 * 16 + lrow;
  const float bo1v = (wave < 8) ? bo1[ocol] : 0.f;

  // zero h_{-1} in LDS
  for (int i = tid; i < B_ * PH; i += NTHREADS) {
    h0[i] = (__bf16)0.f;
    h1[i] = (__bf16)0.f;
  }
  __syncthreads();

#pragma unroll 1
  for (int t = 0; t < T_; ++t) {
    // ---- stage x_t into LDS (fp32 -> bf16) -------------------------------
    for (int i = tid; i < B_ * IN_; i += NTHREADS) {
      const int b = i >> 5, k = i & 31;
      xb[b * PX + k] = (__bf16)xg[(size_t)t * (B_ * IN_) + i];
    }
    __syncthreads();

    // ======================= layer 0 ======================================
    // d_pre = x@Wdx0^T (K=32: one wmma k-step) + h0@Wdh0^T (K=256)
    {
      v8f cd[4] = {zero8(), zero8(), zero8(), zero8()};
      {
        v16bf b = load_b(launder(wts + O_WDX0), IN_, ntile, 0, lrow, lhi);
#pragma unroll
        for (int m = 0; m < 4; ++m) {
          v16bf a = load_a(xb, PX, m, 0, lrow, lhi);
          cd[m] = wmma_bf16(a, b, cd[m]);
        }
      }
      gemm_acc<8>(cd, h0, PH, wts + O_WDH0, H_, ntile, lrow, lhi);
      // d, fractional-memory sum m0 -> mb (lane-private history reads)
      d_and_memory(cd, bd0v, buf0, t, lhi, hcol, mb);
    }
    __syncthreads();

    // h = tanh(x@Wx0^T + m0@Wh0^T + bh0)
    {
      v8f ch[4] = {zero8(), zero8(), zero8(), zero8()};
      {
        v16bf b = load_b(launder(wts + O_WX0), IN_, ntile, 0, lrow, lhi);
#pragma unroll
        for (int m = 0; m < 4; ++m) {
          v16bf a = load_a(xb, PX, m, 0, lrow, lhi);
          ch[m] = wmma_bf16(a, b, ch[m]);
        }
      }
      gemm_acc<8>(ch, mb, PH, wts + O_WH0, H_, ntile, lrow, lhi);
#pragma unroll
      for (int m = 0; m < 4; ++m)
#pragma unroll
        for (int i = 0; i < 8; ++i) {
          const int brow = m * 16 + lhi * 8 + i;
          const float hv = tanhf(ch[m][i] + bh0v);
          h0[brow * PH + hcol] = (__bf16)hv;  // all GEMM1 reads of h0 done
          buf0[(size_t)((t & 15) * (B_ * H_) + brow * H_ + hcol)] = (__bf16)hv;
        }
    }
    __syncthreads();

    // o0 = h@Wo0^T + bo0 -> reuse mb (all mb readers passed previous barrier)
    {
      v8f co[4] = {zero8(), zero8(), zero8(), zero8()};
      gemm_acc<8>(co, h0, PH, wts + O_WO0, H_, ntile, lrow, lhi);
#pragma unroll
      for (int m = 0; m < 4; ++m)
#pragma unroll
        for (int i = 0; i < 8; ++i)
          mb[(m * 16 + lhi * 8 + i) * PH + hcol] = (__bf16)(co[m][i] + bo0v);
    }
    __syncthreads();

    // ======================= layer 1 ======================================
    // d_pre1 = o0@Wdx1^T + h1@Wdh1^T ; m1 -> its own LDS buffer
    {
      v8f cd1[4] = {zero8(), zero8(), zero8(), zero8()};
      gemm_acc<8>(cd1, mb, PH, wts + O_WDX1, H_, ntile, lrow, lhi);
      gemm_acc<8>(cd1, h1, PH, wts + O_WDH1, H_, ntile, lrow, lhi);
      d_and_memory(cd1, bd1v, buf1, t, lhi, hcol, m1b);
    }
    __syncthreads();

    // h1 = tanh(o0@Wx1^T + m1@Wh1^T + bh1)
    {
      v8f cx1[4] = {zero8(), zero8(), zero8(), zero8()};
      gemm_acc<8>(cx1, mb, PH, wts + O_WX1, H_, ntile, lrow, lhi);
      gemm_acc<8>(cx1, m1b, PH, wts + O_WH1, H_, ntile, lrow, lhi);
#pragma unroll
      for (int m = 0; m < 4; ++m)
#pragma unroll
        for (int i = 0; i < 8; ++i) {
          const int brow = m * 16 + lhi * 8 + i;
          const float hv = tanhf(cx1[m][i] + bh1v);
          h1[brow * PH + hcol] = (__bf16)hv;
          buf1[(size_t)((t & 15) * (B_ * H_) + brow * H_ + hcol)] = (__bf16)hv;
        }
    }
    __syncthreads();

    // o1 = h1@Wo1^T + bo1 -> d_out[t]   (8 tiles: waves 0..7, one tile each)
    if (wave < 8) {
      const __bf16* Wo1l = launder(wts + O_WO1);
      v8f c = zero8();
#pragma unroll
      for (int ks = 0; ks < 8; ++ks) {
        v16bf b = load_b(Wo1l, H_, nt1, ks * 32, lrow, lhi);
        v16bf a = load_a(h1, PH, om, ks * 32, lrow, lhi);
        c = wmma_bf16(a, b, c);
      }
#pragma unroll
      for (int i = 0; i < 8; ++i) {
        const int brow = om * 16 + lhi * 8 + i;
        out[(size_t)t * (B_ * OUT_) + brow * OUT_ + ocol] = c[i] + bo1v;
      }
    }
    // next iteration's xb-staging barrier provides the end-of-step sync
  }
}

// ---------------------------------------------------------------------------
__global__ void f32_to_bf16_k(const float* __restrict__ s, __bf16* __restrict__ d,
                              int n) {
  const int i = blockIdx.x * blockDim.x + threadIdx.x;
  if (i < n) d[i] = (__bf16)s[i];
}

__global__ void zero32_k(uint32_t* __restrict__ p, int n) {
  const int i = blockIdx.x * blockDim.x + threadIdx.x;
  if (i < n) p[i] = 0u;
}

// ---------------------------------------------------------------------------
extern "C" void kernel_launch(void* const* d_in, const int* in_sizes, int n_in,
                              void* d_out, int out_size, void* d_ws, size_t ws_size,
                              hipStream_t stream) {
  (void)in_sizes; (void)n_in; (void)out_size; (void)ws_size;
  const float* x    = (const float*)d_in[0];
  const float* Wdx0 = (const float*)d_in[1];
  const float* Wdh0 = (const float*)d_in[2];
  const float* bd0  = (const float*)d_in[3];
  const float* Wx0  = (const float*)d_in[4];
  const float* Wh0  = (const float*)d_in[5];
  const float* bh0  = (const float*)d_in[6];
  const float* Wo0  = (const float*)d_in[7];
  const float* bo0  = (const float*)d_in[8];
  const float* Wdx1 = (const float*)d_in[9];
  const float* Wdh1 = (const float*)d_in[10];
  const float* bd1  = (const float*)d_in[11];
  const float* Wx1  = (const float*)d_in[12];
  const float* Wh1  = (const float*)d_in[13];
  const float* bh1  = (const float*)d_in[14];
  const float* Wo1  = (const float*)d_in[15];
  const float* bo1  = (const float*)d_in[16];

  __bf16* wts  = (__bf16*)d_ws;
  __bf16* buf0 = (__bf16*)((char*)d_ws + BUF_BYTE_OFF);     // [16,B,H] bf16
  __bf16* buf1 = buf0 + 16 * B_ * H_;

  auto cvt = [&](const float* s, int off, int n) {
    f32_to_bf16_k<<<(n + 255) / 256, 256, 0, stream>>>(s, wts + off, n);
  };
  cvt(Wdx0, O_WDX0, H_ * IN_);
  cvt(Wx0,  O_WX0,  H_ * IN_);
  cvt(Wdh0, O_WDH0, H_ * H_);
  cvt(Wh0,  O_WH0,  H_ * H_);
  cvt(Wo0,  O_WO0,  H_ * H_);
  cvt(Wdx1, O_WDX1, H_ * H_);
  cvt(Wdh1, O_WDH1, H_ * H_);
  cvt(Wx1,  O_WX1,  H_ * H_);
  cvt(Wh1,  O_WH1,  H_ * H_);
  cvt(Wo1,  O_WO1,  OUT_ * H_);

  // zero both history buffers (2 * 16*B*H bf16 = 262144 dwords)
  const int nz = (2 * 16 * B_ * H_ * (int)sizeof(__bf16)) / 4;
  zero32_k<<<(nz + 511) / 512, 512, 0, stream>>>((uint32_t*)buf0, nz);

  mrnn_persistent<<<1, NTHREADS, 0, stream>>>(x, bd0, bh0, bo0, bd1, bh1, bo1,
                                              wts, buf0, buf1, (float*)d_out);
}
